// NodalAttention_64493228917345
// MI455X (gfx1250) — matrix operand order repaired and checked by reading, not compiled, not run on hardware
//
#include <hip/hip_runtime.h>
#include <math.h>

typedef __attribute__((ext_vector_type(16))) _Float16 v16h;
typedef __attribute__((ext_vector_type(8)))  float    v8f;

#define MM 256
#define DD 256
#define DEe 64

__device__ __forceinline__ v8f wmma_f16(v16h a, v16h b, v8f c) {
  return __builtin_amdgcn_wmma_f32_16x16x32_f16(false, a, false, b, (short)0, c,
                                                false, false);
}

// ---------------------------------------------------------------------------
// Generic C = A * W^T (+bias), fp32 in/out, f16 WMMA inner product.
// A row r lives at A + (r + (r>>8)*rowPad + rowOff)*lda  (handles the
// features[:,1:] per-batch offset).  W is row-major (N x ldw), k starts at
// wColOff.  One wave per 16x16 output tile, 8 waves per block.
// ---------------------------------------------------------------------------
__global__ __launch_bounds__(256) void gemm_awt(
    const float* __restrict__ A, int lda, int rowPad, int rowOff,
    const float* __restrict__ W, int ldw, int wColOff,
    const float* __restrict__ bias,
    float* __restrict__ C, int ldc,
    int ntilesN, int K)
{
  int tid  = threadIdx.x;
  int wave = tid >> 5, lane = tid & 31;
  int tile = blockIdx.x * 8 + wave;
  int mt = tile / ntilesN, nt = tile % ntilesN;
  int laneN  = lane & 15;
  int kbaseA = (lane < 16) ? 0 : 8;    // A frag K base (ISA 16-bit A layout)
  int kbaseB = (lane < 16) ? 0 : 8;    // B frag K base
  int rowSel = (lane < 16) ? 0 : 8;    // C row offset

  int r = mt * 16 + laneN;
  const float* arow = A + ((size_t)r + (size_t)(r >> 8) * rowPad + rowOff) * lda;
  const float* wrow = W + (size_t)(nt * 16 + laneN) * ldw + wColOff;

  v8f c = {0.f, 0.f, 0.f, 0.f, 0.f, 0.f, 0.f, 0.f};
  for (int k0 = 0; k0 < K; k0 += 32) {
    v16h a, b;
#pragma unroll
    for (int v = 0; v < 8; ++v) {
      int ea = k0 + kbaseA + ((v < 4) ? 2 * v : 16 + 2 * (v - 4));
      a[2 * v]     = (_Float16)arow[ea];
      a[2 * v + 1] = (_Float16)arow[ea + 1];
      int eb = k0 + kbaseB + ((v < 4) ? 2 * v : 16 + 2 * (v - 4));
      b[2 * v]     = (_Float16)wrow[eb];
      b[2 * v + 1] = (_Float16)wrow[eb + 1];
    }
    c = wmma_f16(a, b, c);
  }
  float bv = bias ? bias[nt * 16 + laneN] : 0.f;
#pragma unroll
  for (int rr = 0; rr < 8; ++rr) {
    int m = mt * 16 + rr + rowSel;
    C[(size_t)m * ldc + nt * 16 + laneN] = c[rr] + bv;
  }
}

// ---------------------------------------------------------------------------
// fwnf[b,j] = f[b,j,:].wa_nf ; aspwa[i] = f[0,i,:].wa_asp
// ---------------------------------------------------------------------------
__global__ __launch_bounds__(256) void vec_dots(const float* __restrict__ f,
                                                const float* __restrict__ Wa,
                                                float* __restrict__ fwnf,
                                                float* __restrict__ aspwa)
{
  int gid = blockIdx.x * 256 + threadIdx.x;
  if (gid < 512) {
    const float* row = f + (size_t)gid * DD;
    float acc = 0.f;
    for (int c = 0; c < DD; ++c) acc += row[c] * Wa[DEe + c];
    fwnf[gid] = acc;
  } else if (gid < 768) {
    int i = gid - 512;
    const float* row = f + (size_t)i * DD;   // f[0, i]
    float acc = 0.f;
    for (int c = 0; c < DD; ++c) acc += row[c] * Wa[DEe + DD + c];
    aspwa[i] = acc;
  }
}

// ---------------------------------------------------------------------------
// One block per (b, t) with i = asp_start[b] + t.  Computes masked-softmax
// attention row w[0..511] then fused[b*16+t, :] = sum_k w_k * relu(msg_k).
// ---------------------------------------------------------------------------
__global__ __launch_bounds__(256) void monster(
    const float* __restrict__ dep, const int* __restrict__ adj,
    const int* __restrict__ asp_start,
    const float* __restrict__ Wa, const float* __restrict__ Wm,
    const float* __restrict__ fwnf, const float* __restrict__ aspwa,
    const float* __restrict__ nfp, float* __restrict__ fused)
{
  __shared__ float s_w[512];
  __shared__ float s_red[256];
  __shared__ float s_wad[DEe];

  int tid = threadIdx.x;
  int b = blockIdx.x >> 4, t = blockIdx.x & 15;
  int i = asp_start[b] + t;

  if (tid < DEe) s_wad[tid] = Wa[tid];
  __syncthreads();

  // ---- phase 1: s[k] (leaky-relu, masked) -------------------------------
  {
    const float* dv0 = dep + (((size_t)b * MM + i) * MM + tid) * DEe;
    int m0 = adj[((size_t)b * MM + i) * MM + tid];
    float a0 = 0.f;
    for (int e = 0; e < DEe; ++e) a0 += dv0[e] * s_wad[e];
    float sv0 = a0 + fwnf[b * MM + tid] + aspwa[i];
    sv0 = sv0 > 0.f ? sv0 : 0.01f * sv0;
    s_w[tid] = m0 ? sv0 : -3.402823466e38f;

    const float* dv1 = dep + (((size_t)b * MM + tid) * MM + i) * DEe;
    int m1 = adj[((size_t)b * MM + tid) * MM + i];
    float a1 = 0.f;
    for (int e = 0; e < DEe; ++e) a1 += dv1[e] * s_wad[e];
    float sv1 = a1 + fwnf[b * MM + tid] + aspwa[i];
    sv1 = sv1 > 0.f ? sv1 : 0.01f * sv1;
    s_w[tid + 256] = m1 ? sv1 : -3.402823466e38f;
  }
  __syncthreads();

  // ---- softmax over 512 entries ----------------------------------------
  s_red[tid] = fmaxf(s_w[tid], s_w[tid + 256]);
  __syncthreads();
  for (int off = 128; off > 0; off >>= 1) {
    if (tid < off) s_red[tid] = fmaxf(s_red[tid], s_red[tid + off]);
    __syncthreads();
  }
  float mx = s_red[0];
  __syncthreads();
  float e0 = expf(s_w[tid] - mx);        // masked entries -> exactly 0
  float e1 = expf(s_w[tid + 256] - mx);
  s_red[tid] = e0 + e1;
  __syncthreads();
  for (int off = 128; off > 0; off >>= 1) {
    if (tid < off) s_red[tid] = s_red[tid] + s_red[tid + off];
    __syncthreads();
  }
  float inv = 1.0f / s_red[0];
  s_w[tid]       = e0 * inv;
  s_w[tid + 256] = e1 * inv;
  __syncthreads();

  // ---- phase 2: fused[d] = sum_k w[k]*relu(dep_cat[k].Wm_dep^T + nfp) ---
  int wave = tid >> 5, lane = tid & 31;
  int laneN  = lane & 15;
  int kbaseA = (lane < 16) ? 0 : 8;
  int kbaseB = (lane < 16) ? 0 : 8;
  int rowSel = (lane < 16) ? 0 : 8;
  size_t depRowI = ((size_t)b * MM + i) * MM;

  for (int q = 0; q < 2; ++q) {
    int d0 = (wave * 2 + q) * 16;
    const float* wrow = Wm + (size_t)(d0 + laneN) * (DEe + DD);  // Wm_dep cols
    v16h bf0, bf1;
#pragma unroll
    for (int v = 0; v < 8; ++v) {
      int e = kbaseB + ((v < 4) ? 2 * v : 16 + 2 * (v - 4));
      bf0[2 * v]     = (_Float16)wrow[e];
      bf0[2 * v + 1] = (_Float16)wrow[e + 1];
      bf1[2 * v]     = (_Float16)wrow[32 + e];
      bf1[2 * v + 1] = (_Float16)wrow[32 + e + 1];
    }
    float part = 0.f;
    for (int kt = 0; kt < 32; ++kt) {
      int krow = kt * 16 + laneN;
      const float* dv = (krow < MM)
          ? dep + (depRowI + krow) * DEe
          : dep + (((size_t)b * MM + (krow - MM)) * MM + i) * DEe;
      v16h a0, a1;
#pragma unroll
      for (int v = 0; v < 8; ++v) {
        int e = kbaseA + ((v < 4) ? 2 * v : 16 + 2 * (v - 4));
        a0[2 * v]     = (_Float16)dv[e];
        a0[2 * v + 1] = (_Float16)dv[e + 1];
        a1[2 * v]     = (_Float16)dv[32 + e];
        a1[2 * v + 1] = (_Float16)dv[32 + e + 1];
      }
      v8f c = {0.f, 0.f, 0.f, 0.f, 0.f, 0.f, 0.f, 0.f};
      c = wmma_f16(a0, bf0, c);
      c = wmma_f16(a1, bf1, c);
#pragma unroll
      for (int rr = 0; rr < 8; ++rr) {
        int kc   = kt * 16 + rr + rowSel;
        int kmod = kc & (MM - 1);
        float msg = c[rr] + nfp[((size_t)b * MM + kmod) * DD + d0 + laneN];
        msg = msg > 0.f ? msg : 0.f;
        part += s_w[kc] * msg;
      }
    }
    float other = __shfl_xor(part, 16);
    part += other;
    if (lane < 16) fused[((size_t)b * 16 + t) * DD + d0 + laneN] = part;
  }
}

// ---------------------------------------------------------------------------
__global__ void copy_out(const float* __restrict__ feat, float* __restrict__ out,
                         int n)
{
  int idx = blockIdx.x * 256 + threadIdx.x;
  if (idx < n) out[idx] = feat[idx];
}

__global__ __launch_bounds__(256) void out_rows(
    const float* __restrict__ fused, const float* __restrict__ f,
    const float* __restrict__ Wh, const int* __restrict__ asp_start,
    float* __restrict__ out)
{
  int b = blockIdx.x >> 4, t = blockIdx.x & 15;
  int i = asp_start[b] + t;
  int d = threadIdx.x;
  const float* fr = fused + ((size_t)b * 16 + t) * DD;
  const float* ar = f + (size_t)i * DD;            // asp[i] = f[0, i]
  const float* wh = Wh + (size_t)d * (2 * DD);
  float acc = 0.f;
  for (int c = 0; c < DD; ++c) acc += fr[c] * wh[c];
  for (int c = 0; c < DD; ++c) acc += ar[c] * wh[DD + c];
  acc = acc > 0.f ? acc : 0.f;
  out[((size_t)b * (MM + 1) + 1 + i) * DD + d] = acc;
}

// ---------------------------------------------------------------------------
extern "C" void kernel_launch(void* const* d_in, const int* in_sizes, int n_in,
                              void* d_out, int out_size, void* d_ws, size_t ws_size,
                              hipStream_t stream)
{
  const float* features  = (const float*)d_in[0];
  const float* dep       = (const float*)d_in[1];
  const int*   adj       = (const int*)d_in[2];
  const int*   asp_start = (const int*)d_in[3];
  /* d_in[4] = asp_end (== asp_start + 16, unused) */
  const float* Wz = (const float*)d_in[5];
  const float* bz = (const float*)d_in[6];
  const float* Wa = (const float*)d_in[7];
  const float* Wm = (const float*)d_in[8];
  const float* Wh = (const float*)d_in[9];
  float* out = (float*)d_out;

  float* ws    = (float*)d_ws;
  float* f     = ws;               // (2,256,256)
  float* nfp   = ws + 131072;      // (2,256,256)
  float* fwnf  = ws + 262144;      // (2,256)
  float* aspwa = ws + 262656;      // (256,)
  float* fusedv = ws + 262912;     // (32,256)

  // f = features[:,1:] @ Wz^T + bz   (row map skips the leading token per b)
  gemm_awt<<<64, 256, 0, stream>>>(features, DD, 1, 1, Wz, DD, 0, bz,
                                   f, DD, 16, DD);
  vec_dots<<<3, 256, 0, stream>>>(f, Wa, fwnf, aspwa);
  // nfp = f @ Wm_f^T   (Wm columns 64..319)
  gemm_awt<<<64, 256, 0, stream>>>(f, DD, 0, 0, Wm, DEe + DD, DEe, nullptr,
                                   nfp, DD, 16, DD);
  monster<<<32, 256, 0, stream>>>(dep, adj, asp_start, Wa, Wm, fwnf, aspwa,
                                  nfp, fusedv);
  copy_out<<<(2 * (MM + 1) * DD + 255) / 256, 256, 0, stream>>>(
      features, out, 2 * (MM + 1) * DD);
  out_rows<<<32, 256, 0, stream>>>(fusedv, f, Wh, asp_start, out);
}
